// CPELoss_53352083751397
// MI455X (gfx1250) — matrix operand order repaired and checked by reading, not compiled, not run on hardware
//
#include <hip/hip_runtime.h>
#include <hip/hip_bf16.h>
#include <math.h>

// ---------------------------------------------------------------------------
// CPE contrastive loss on MI455X (gfx1250), wave32 + WMMA f16->f32.
//   N = 8192 proposals, D = 256 features.
//   sim = (f/||f||) . (f/||f||)^T / 0.1  -> masked softmax loss -> scalar.
// Logits are bounded (|s| <= ~10.3), so exp(s) is accumulated directly with
// no running max: the reference's clips / loss cap make the result identical.
// ---------------------------------------------------------------------------

#define N_PROP 8192
#define D_FEAT 256
#define NJ 8                 // j-dimension chunks (parallelism for partials)
#define JCHUNK (N_PROP / NJ) // 1024 columns per chunk
#define JT (JCHUNK / 16)     // 64 j-tiles per chunk
#define RS 264               // padded LDS row stride in halves (4-bank shift/row)

typedef _Float16 v16h __attribute__((ext_vector_type(16)));
typedef _Float16 v8h  __attribute__((ext_vector_type(8)));
typedef float    v8f  __attribute__((ext_vector_type(8)));
typedef int      v4i  __attribute__((vector_size(16)));

#if defined(__has_builtin)
#if __has_builtin(__builtin_amdgcn_global_load_async_to_lds_b128)
#define CPE_HAVE_ASYNC 1
#endif
#endif

// ---------------------------------------------------------------------------
// Kernel 1: row-normalize (f32) -> f16 copy in workspace; zero accumulators.
// ---------------------------------------------------------------------------
__global__ void cpe_normalize_kernel(const float* __restrict__ feat,
                                     _Float16* __restrict__ fh,
                                     float* __restrict__ acc) {
    const int row = blockIdx.x;
    const int tid = threadIdx.x;
    const float x = feat[(size_t)row * D_FEAT + tid];

    float v = x * x;
    #pragma unroll
    for (int off = 16; off > 0; off >>= 1) v += __shfl_xor(v, off, 32);

    __shared__ float red[8];
    const int lane = tid & 31, wid = tid >> 5;
    if (lane == 0) red[wid] = v;
    __syncthreads();
    if (tid == 0) {
        float s = 0.f;
        #pragma unroll
        for (int k = 0; k < 8; ++k) s += red[k];
        red[0] = s;
    }
    __syncthreads();
    const float inv = 1.0f / fmaxf(sqrtf(red[0]), 1e-12f);
    fh[(size_t)row * D_FEAT + tid] = (_Float16)(x * inv);

    if (blockIdx.x == 0 && tid < 2) acc[tid] = 0.0f; // total, count
}

// ---------------------------------------------------------------------------
// Kernel 2: tiled f16 WMMA similarity + masked exp-sum accumulation.
// Block = 256 threads = 8 waves; wave w owns i-tile (iBlock*8+w)*16 rows; all
// waves share j-chunk, staging 16x256 f16 j-tiles in LDS (async copy path).
// ---------------------------------------------------------------------------
__global__ __launch_bounds__(256) void
cpe_sim_lse_kernel(const _Float16* __restrict__ fh,
                   const float* __restrict__ ious,
                   const int* __restrict__ labels,
                   float* __restrict__ pps, float* __restrict__ pas,
                   float* __restrict__ pap, float* __restrict__ pan) {
    __shared__ _Float16 tile[16 * RS]; // 8448 B staged j-tile

    const int iBlock = blockIdx.x / NJ;
    const int chunk  = blockIdx.x % NJ;
    const int wave   = threadIdx.x >> 5;
    const int lane   = threadIdx.x & 31;
    const int n      = lane & 15;   // C column / B lane-column
    const int hi     = lane >> 4;   // half-wave id
    const int ibase  = (iBlock * 8 + wave) * 16;

    // ---- Preload A fragments: 16 rows x K=256, 8 slabs of K=32 ----
    // ISA 16-bit A layout: lanes 0-15 -> k[ks..ks+7] & k[ks+16..ks+23],
    //                      lanes 16-31 -> k[ks+8..ks+15] & k[ks+24..ks+31].
    v16h afrag[8];
    {
        const v8h* rowp = (const v8h*)(fh + (size_t)(ibase + n) * D_FEAT);
        #pragma unroll
        for (int s = 0; s < 8; ++s) {
            v8h lo  = rowp[4 * s + hi];
            v8h hi8 = rowp[4 * s + 2 + hi];
            afrag[s] = __builtin_shufflevector(lo, hi8,
                0,1,2,3,4,5,6,7,8,9,10,11,12,13,14,15);
        }
    }

    // ---- Per-lane i-side mask data for rows M = v + 8*hi ----
    float ii[8]; int li[8]; int irow[8]; int ipos[8], ineg[8], ifg[8];
    #pragma unroll
    for (int v = 0; v < 8; ++v) {
        irow[v] = ibase + v + 8 * hi;
        ii[v]   = ious[irow[v]];
        li[v]   = labels[irow[v]];
        ipos[v] = (ii[v] > 0.5f);
        ineg[v] = (ii[v] < 0.3f);
        ifg[v]  = (li[v] >= 0);
    }

    float ps[8], as_[8];
    #pragma unroll
    for (int v = 0; v < 8; ++v) { ps[v] = 0.f; as_[v] = 0.f; }
    int apos = 0, aneg = 0;

    const int jcb  = chunk * JCHUNK;
    const int srow = threadIdx.x >> 4;   // staging row 0..15
    const int sseg = threadIdx.x & 15;   // staging 16-half segment

    for (int jt = 0; jt < JT; ++jt) {
        const int jb = jcb + jt * 16;

        // ---- Stage j-tile (16 rows x 256 halves) into padded LDS ----
        __syncthreads();
        {
            const _Float16* gsrc = fh + (size_t)(jb + srow) * D_FEAT + sseg * 16;
            _Float16* ldst = tile + srow * RS + sseg * 16;
#if CPE_HAVE_ASYNC
            __builtin_amdgcn_global_load_async_to_lds_b128(
                (v4i*)gsrc, (v4i*)ldst, 0, 0);
            __builtin_amdgcn_global_load_async_to_lds_b128(
                (v4i*)(gsrc + 8), (v4i*)(ldst + 8), 0, 0);
#if __has_builtin(__builtin_amdgcn_s_wait_asynccnt)
            __builtin_amdgcn_s_wait_asynccnt(0);
#else
            asm volatile("s_wait_asynccnt 0" ::: "memory");
#endif
#else
            const uint4* g = (const uint4*)gsrc;
            uint4* l = (uint4*)ldst;
            l[0] = g[0];
            l[1] = g[1];
#endif
            // prefetch next tile (wrap to avoid loop peeling)
            const int pjb = (jt + 1 < JT) ? (jb + 16) : jcb;
            __builtin_prefetch(fh + (size_t)(pjb + srow) * D_FEAT + sseg * 16, 0, 1);
        }
        __syncthreads();

        // ---- j-side mask data ----
        const int   lj   = labels[jb + n];
        const float ioj  = ious[jb + n];
        const int   jfg  = (lj >= 0);
        const int   jpos = (ioj > 0.5f);

        // ---- 16x16 sim tile via 8x WMMA f16 K=32 ----
        v8f c = {};
        const v8h* lp = (const v8h*)(tile + n * RS);
        #pragma unroll
        for (int s = 0; s < 8; ++s) {
            v8h blo = lp[4 * s + hi];
            v8h bhi = lp[4 * s + 2 + hi];
            v16h b = __builtin_shufflevector(blo, bhi,
                0,1,2,3,4,5,6,7,8,9,10,11,12,13,14,15);
            c = __builtin_amdgcn_wmma_f32_16x16x32_f16(
                    false, afrag[s], false, b, (short)0, c, false, false);
        }

        // ---- Branchless mask + exp accumulation, elem (M=v+8*hi, N=n) ----
        #pragma unroll
        for (int v = 0; v < 8; ++v) {
            const float s = c[v] * 10.0f;               // / TAU
            const int same   = (li[v] == lj);
            const int close  = (fabsf(ioj - ii[v]) < 0.2f);
            const int pairok = ifg[v] & jfg & (irow[v] != (jb + n));
            const int pos = same & ipos[v] & jpos & close & pairok;
            const int neg = ((same ^ 1) | ineg[v]) & pairok;
            const float e = __expf(s);
            as_[v] += (pos | neg) ? e : 0.0f;
            ps[v]  += pos ? e : 0.0f;
            apos |= pos << v;
            aneg |= neg << v;
        }
    }

    // ---- Sum the 16 lanes of each half-wave (per-row reduce) ----
    #pragma unroll
    for (int off = 8; off >= 1; off >>= 1) {
        #pragma unroll
        for (int v = 0; v < 8; ++v) {
            ps[v]  += __shfl_xor(ps[v],  off, 16);
            as_[v] += __shfl_xor(as_[v], off, 16);
        }
        apos |= __shfl_xor(apos, off, 16);
        aneg |= __shfl_xor(aneg, off, 16);
    }

    // ---- Lanes 0 / 16 write partial state for their 8 rows ----
    if (n == 0) {
        #pragma unroll
        for (int v = 0; v < 8; ++v) {
            const int idx = irow[v] * NJ + chunk;
            pps[idx] = ps[v];
            pas[idx] = as_[v];
            pap[idx] = ((apos >> v) & 1) ? 1.0f : 0.0f;
            pan[idx] = ((aneg >> v) & 1) ? 1.0f : 0.0f;
        }
    }
}

// ---------------------------------------------------------------------------
// Kernel 3: merge NJ chunk sums per row, compute loss_i, accumulate.
// ---------------------------------------------------------------------------
__global__ void cpe_merge_kernel(const int* __restrict__ labels,
                                 const float* __restrict__ pps,
                                 const float* __restrict__ pas,
                                 const float* __restrict__ pap,
                                 const float* __restrict__ pan,
                                 float* __restrict__ acc) {
    const int row = blockIdx.x * blockDim.x + threadIdx.x;
    if (row >= N_PROP) return;

    float ps = 0.f, as_ = 0.f, ap = 0.f, an = 0.f;
    #pragma unroll
    for (int c = 0; c < NJ; ++c) {
        const int idx = row * NJ + c;
        ps  += pps[idx];
        as_ += pas[idx];
        ap  += pap[idx];
        an  += pan[idx];
    }

    const bool valid = (labels[row] >= 0) && (ap > 0.f) && (an > 0.f);
    if (valid) {
        // Shift-invariant: clips in the reference are no-ops for valid rows
        // given the min(loss, 10) cap.
        const float loss = fminf(logf(as_) - logf(ps), 10.0f);
        atomicAdd(&acc[0], loss);
        atomicAdd(&acc[1], 1.0f);
    }
}

// ---------------------------------------------------------------------------
// Kernel 4: finalize scalar.
// ---------------------------------------------------------------------------
__global__ void cpe_final_kernel(const float* __restrict__ acc,
                                 float* __restrict__ out) {
    out[0] = (acc[1] > 0.5f) ? (acc[0] / acc[1]) : 0.0f;
}

// ---------------------------------------------------------------------------
extern "C" void kernel_launch(void* const* d_in, const int* in_sizes, int n_in,
                              void* d_out, int out_size, void* d_ws, size_t ws_size,
                              hipStream_t stream) {
    (void)in_sizes; (void)n_in; (void)out_size; (void)ws_size;
    const float* feat   = (const float*)d_in[0];
    const float* ious   = (const float*)d_in[1];
    const int*   labels = (const int*)d_in[2];
    float* out = (float*)d_out;

    char* ws = (char*)d_ws;
    _Float16* fh = (_Float16*)ws;                                  // 4 MB f16 features
    float* pps = (float*)(ws + (size_t)N_PROP * D_FEAT * 2);       // 4 x 256 KB partials
    float* pas = pps + N_PROP * NJ;
    float* pap = pas + N_PROP * NJ;
    float* pan = pap + N_PROP * NJ;
    float* acc = pan + N_PROP * NJ;                                // [total, count]

    cpe_normalize_kernel<<<dim3(N_PROP), dim3(D_FEAT), 0, stream>>>(feat, fh, acc);
    cpe_sim_lse_kernel<<<dim3((N_PROP / 16 / 8) * NJ), dim3(256), 0, stream>>>(
        fh, ious, labels, pps, pas, pap, pan);
    cpe_merge_kernel<<<dim3(N_PROP / 256), dim3(256), 0, stream>>>(
        labels, pps, pas, pap, pan, acc);
    cpe_final_kernel<<<dim3(1), dim3(1), 0, stream>>>(acc, out);
}
